// LinearRegressionMaskNorm_53919019433997
// MI455X (gfx1250) — matrix operand order repaired and checked by reading, not compiled, not run on hardware
//
#include <hip/hip_runtime.h>
#include <hip/hip_bf16.h>

typedef __attribute__((ext_vector_type(16))) __bf16 v16bf;
typedef __attribute__((ext_vector_type(8)))  __bf16 v8bf;
typedef __attribute__((ext_vector_type(8)))  float  v8f;

#define TILE      128
#define KCHUNK    32
#define LDS_PITCH 40   // ushorts per row: 32 data + 8 pad (20 banks -> conflict-free)
#define ARRSZ     (TILE * LDS_PITCH)   // ushorts per staged tile array

// Split fp32 into bf16 hi (truncation) + bf16 lo (RNE of residual).
__device__ __forceinline__ void split_bf16(float x, unsigned short& h, unsigned short& l) {
    unsigned u  = __builtin_bit_cast(unsigned, x);
    unsigned hu = u & 0xFFFF0000u;
    h = (unsigned short)(hu >> 16);
    float r = x - __builtin_bit_cast(float, hu);
    unsigned v = __builtin_bit_cast(unsigned, r);
    v += 0x7FFFu + ((v >> 16) & 1u);
    l = (unsigned short)(v >> 16);
}

__device__ __forceinline__ unsigned long long pack4(unsigned short a, unsigned short b,
                                                    unsigned short c, unsigned short d) {
    return (unsigned long long)a | ((unsigned long long)b << 16) |
           ((unsigned long long)c << 32) | ((unsigned long long)d << 48);
}

#define SHUF16(a, b) __builtin_shufflevector(a, b, 0,1,2,3,4,5,6,7,8,9,10,11,12,13,14,15)

// Async global->LDS 16-byte copy (CDNA5, ASYNCcnt-tracked).
__device__ __forceinline__ void async_b128(unsigned lds_addr, const void* gbase, unsigned goff) {
    asm volatile("global_load_async_to_lds_b128 %0, %1, %2"
                 :: "v"(lds_addr), "v"(goff), "s"(gbase) : "memory");
}

// ---------------------------------------------------------------------------
// Pre-pass: fp32 -> (bf16 hi, bf16 lo) arrays, vectorized x4.
// ---------------------------------------------------------------------------
__global__ __launch_bounds__(256) void split_kernel(
    const float* __restrict__ src,
    unsigned short* __restrict__ hi, unsigned short* __restrict__ lo, long n4)
{
    long i = (long)blockIdx.x * 256 + threadIdx.x;
    if (i >= n4) return;
    float4 v = ((const float4*)src)[i];
    unsigned short h0,h1,h2,h3,l0,l1,l2,l3;
    split_bf16(v.x, h0, l0); split_bf16(v.y, h1, l1);
    split_bf16(v.z, h2, l2); split_bf16(v.w, h3, l3);
    ((unsigned long long*)hi)[i] = pack4(h0,h1,h2,h3);
    ((unsigned long long*)lo)[i] = pack4(l0,l1,l2,l3);
}

// ---------------------------------------------------------------------------
// Main GEMM (pre-split path): bf16 hi/lo tiles staged via async-to-LDS,
// double-buffered; 24 bf16 WMMAs per K-chunk per wave (bf16x3 emulated fp32).
// ---------------------------------------------------------------------------
__global__ __launch_bounds__(256) void gemm_mask_pre_kernel(
    const unsigned short* __restrict__ Xh, const unsigned short* __restrict__ Xl,
    const unsigned short* __restrict__ Ah, const unsigned short* __restrict__ Al,
    const float* __restrict__ X, const float* __restrict__ bias,
    float* __restrict__ out, int N, int D)
{
    // [buffer][array: Xh,Xl,Ah,Al][row*pitch]
    __shared__ __align__(16) unsigned short lds[2][4][ARRSZ];

    const int c0   = blockIdx.x * TILE;
    const int n0   = blockIdx.y * TILE;
    const int t    = threadIdx.x;
    const int lane = t & 31;
    const int wave = t >> 5;
    const int wm   = wave & 3;
    const int wn   = wave >> 2;
    const int KC   = D / KCHUNK;

    const v8f vzero = {0.f,0.f,0.f,0.f,0.f,0.f,0.f,0.f};
    v8f acc[2][4];
#pragma unroll
    for (int mi = 0; mi < 2; ++mi)
#pragma unroll
        for (int nj = 0; nj < 4; ++nj) acc[mi][nj] = vzero;

    // Issue the 8 async b128 copies this thread owns for one K-chunk.
    auto stage = [&](int buf, int kc) {
        const int kcol = kc * KCHUNK;
        unsigned lbase = (unsigned)(size_t)(&lds[buf][0][0]);
#pragma unroll
        for (int j = 0; j < 2; ++j) {
            int seg = j * 256 + t;          // 512 16B-segments per tile array
            int row = seg >> 2, q = seg & 3;
            unsigned lq = (unsigned)(row * (LDS_PITCH * 2) + q * 16);
            unsigned gx = (unsigned)(((size_t)(n0 + row) * D + kcol) * 2 + q * 16);
            unsigned ga = (unsigned)(((size_t)(c0 + row) * D + kcol) * 2 + q * 16);
            async_b128(lbase + 0 * (ARRSZ * 2) + lq, Xh, gx);
            async_b128(lbase + 1 * (ARRSZ * 2) + lq, Xl, gx);
            async_b128(lbase + 2 * (ARRSZ * 2) + lq, Ah, ga);
            async_b128(lbase + 3 * (ARRSZ * 2) + lq, Al, ga);
        }
    };

    stage(0, 0);

    for (int kc = 0; kc < KC; ++kc) {
        const int cur = kc & 1;
        if (kc + 1 < KC) {
            stage(cur ^ 1, kc + 1);                          // prefetch next chunk
            asm volatile("s_wait_asynccnt 0x8" ::: "memory"); // chunk kc's 8 done
        } else {
            asm volatile("s_wait_asynccnt 0x0" ::: "memory");
        }
        __syncthreads();

        const unsigned short* sXh = &lds[cur][0][0];
        const unsigned short* sXl = &lds[cur][1][0];
        const unsigned short* sAh = &lds[cur][2][0];
        const unsigned short* sAl = &lds[cur][3][0];

        // A-operand fragments (rows of X): lanes<16 K 0..7 & 16..23, lanes>=16 K 8..15 & 24..31.
        v16bf fXh[2], fXl[2];
        const int kbA = (lane < 16) ? 0 : 8;
#pragma unroll
        for (int mi = 0; mi < 2; ++mi) {
            int row = (wm * 2 + mi) * 16 + (lane & 15);
            int off = row * LDS_PITCH + kbA;
            v8bf a0 = *(const v8bf*)(sXh + off);
            v8bf a1 = *(const v8bf*)(sXh + off + 16);
            fXh[mi] = SHUF16(a0, a1);
            v8bf b0 = *(const v8bf*)(sXl + off);
            v8bf b1 = *(const v8bf*)(sXl + off + 16);
            fXl[mi] = SHUF16(b0, b1);
        }
        // B-operand fragments: column (lane&15), lanes<16 K 0..7 & 16..23, lanes>=16 K 8..15 & 24..31.
        const int kbB = (lane < 16) ? 0 : 8;
#pragma unroll
        for (int nj = 0; nj < 4; ++nj) {
            int col = (wn * 4 + nj) * 16 + (lane & 15);
            int off = col * LDS_PITCH + kbB;
            v8bf h0 = *(const v8bf*)(sAh + off);
            v8bf h1 = *(const v8bf*)(sAh + off + 16);
            v16bf fBh = SHUF16(h0, h1);
            v8bf q0 = *(const v8bf*)(sAl + off);
            v8bf q1 = *(const v8bf*)(sAl + off + 16);
            v16bf fBl = SHUF16(q0, q1);
#pragma unroll
            for (int mi = 0; mi < 2; ++mi) {
                acc[mi][nj] = __builtin_amdgcn_wmma_f32_16x16x32_bf16(
                    false, fXh[mi], false, fBh, (short)0, acc[mi][nj], false, false);
                acc[mi][nj] = __builtin_amdgcn_wmma_f32_16x16x32_bf16(
                    false, fXh[mi], false, fBl, (short)0, acc[mi][nj], false, false);
                acc[mi][nj] = __builtin_amdgcn_wmma_f32_16x16x32_bf16(
                    false, fXl[mi], false, fBh, (short)0, acc[mi][nj], false, false);
            }
        }
        __syncthreads();
    }

    // Epilogue: +bias, mask by (X != 0), store masked y_pred.
    const int rsel = (lane & 16) ? 8 : 0;
#pragma unroll
    for (int nj = 0; nj < 4; ++nj) {
        int gc = c0 + (wn * 4 + nj) * 16 + (lane & 15);
        float bv = bias[gc];
#pragma unroll
        for (int mi = 0; mi < 2; ++mi) {
            int rb = n0 + (wm * 2 + mi) * 16 + rsel;
#pragma unroll
            for (int v = 0; v < 8; ++v) {
                int gr = rb + v;
                float yp = acc[mi][nj][v] + bv;
                float xm = X[(size_t)gr * D + gc];
                out[(size_t)gr * D + gc] = (xm != 0.0f) ? yp : 0.0f;
            }
        }
    }
}

// ---------------------------------------------------------------------------
// Fallback GEMM (round-1): in-loop fp32->bf16 hi/lo conversion, no workspace.
// ---------------------------------------------------------------------------
__global__ __launch_bounds__(256) void gemm_mask_conv_kernel(
    const float* __restrict__ X, const float* __restrict__ A,
    const float* __restrict__ bias, float* __restrict__ out,
    int N, int D)
{
    __shared__ __align__(16) unsigned short sXh[ARRSZ];
    __shared__ __align__(16) unsigned short sXl[ARRSZ];
    __shared__ __align__(16) unsigned short sAh[ARRSZ];
    __shared__ __align__(16) unsigned short sAl[ARRSZ];

    const int c0   = blockIdx.x * TILE;
    const int n0   = blockIdx.y * TILE;
    const int t    = threadIdx.x;
    const int lane = t & 31;
    const int wave = t >> 5;
    const int wm   = wave & 3;
    const int wn   = wave >> 2;
    const int KC   = D / KCHUNK;

    const v8f vzero = {0.f,0.f,0.f,0.f,0.f,0.f,0.f,0.f};
    v8f acc[2][4];
#pragma unroll
    for (int mi = 0; mi < 2; ++mi)
#pragma unroll
        for (int nj = 0; nj < 4; ++nj) acc[mi][nj] = vzero;

    float4 rx[4], ra[4];
#pragma unroll
    for (int i = 0; i < 4; ++i) {
        int idx = i * 256 + t;
        int row = idx >> 3, c4 = idx & 7;
        rx[i] = *((const float4*)(X + (size_t)(n0 + row) * D) + c4);
        ra[i] = *((const float4*)(A + (size_t)(c0 + row) * D) + c4);
    }

    for (int kc = 0; kc < KC; ++kc) {
        __syncthreads();
#pragma unroll
        for (int i = 0; i < 4; ++i) {
            int idx = i * 256 + t;
            int row = idx >> 3, c4 = idx & 7;
            int off = row * LDS_PITCH + c4 * 4;
            unsigned short h0,h1,h2,h3,l0,l1,l2,l3;
            split_bf16(rx[i].x, h0, l0); split_bf16(rx[i].y, h1, l1);
            split_bf16(rx[i].z, h2, l2); split_bf16(rx[i].w, h3, l3);
            *(unsigned long long*)(sXh + off) = pack4(h0,h1,h2,h3);
            *(unsigned long long*)(sXl + off) = pack4(l0,l1,l2,l3);
            split_bf16(ra[i].x, h0, l0); split_bf16(ra[i].y, h1, l1);
            split_bf16(ra[i].z, h2, l2); split_bf16(ra[i].w, h3, l3);
            *(unsigned long long*)(sAh + off) = pack4(h0,h1,h2,h3);
            *(unsigned long long*)(sAl + off) = pack4(l0,l1,l2,l3);
        }
        __syncthreads();

        if (kc + 1 < KC) {
            const int kcol = (kc + 1) * KCHUNK;
#pragma unroll
            for (int i = 0; i < 4; ++i) {
                int idx = i * 256 + t;
                int row = idx >> 3, c4 = idx & 7;
                rx[i] = *((const float4*)(X + (size_t)(n0 + row) * D + kcol) + c4);
                ra[i] = *((const float4*)(A + (size_t)(c0 + row) * D + kcol) + c4);
            }
        }

        v16bf fXh[2], fXl[2];
        const int kbA = (lane < 16) ? 0 : 8;
#pragma unroll
        for (int mi = 0; mi < 2; ++mi) {
            int row = (wm * 2 + mi) * 16 + (lane & 15);
            int off = row * LDS_PITCH + kbA;
            v8bf a0 = *(const v8bf*)(sXh + off);
            v8bf a1 = *(const v8bf*)(sXh + off + 16);
            fXh[mi] = SHUF16(a0, a1);
            v8bf b0 = *(const v8bf*)(sXl + off);
            v8bf b1 = *(const v8bf*)(sXl + off + 16);
            fXl[mi] = SHUF16(b0, b1);
        }
        const int kbB = (lane < 16) ? 0 : 8;
#pragma unroll
        for (int nj = 0; nj < 4; ++nj) {
            int col = (wn * 4 + nj) * 16 + (lane & 15);
            int off = col * LDS_PITCH + kbB;
            v8bf h0 = *(const v8bf*)(sAh + off);
            v8bf h1 = *(const v8bf*)(sAh + off + 16);
            v16bf fBh = SHUF16(h0, h1);
            v8bf q0 = *(const v8bf*)(sAl + off);
            v8bf q1 = *(const v8bf*)(sAl + off + 16);
            v16bf fBl = SHUF16(q0, q1);
#pragma unroll
            for (int mi = 0; mi < 2; ++mi) {
                acc[mi][nj] = __builtin_amdgcn_wmma_f32_16x16x32_bf16(
                    false, fXh[mi], false, fBh, (short)0, acc[mi][nj], false, false);
                acc[mi][nj] = __builtin_amdgcn_wmma_f32_16x16x32_bf16(
                    false, fXh[mi], false, fBl, (short)0, acc[mi][nj], false, false);
                acc[mi][nj] = __builtin_amdgcn_wmma_f32_16x16x32_bf16(
                    false, fXl[mi], false, fBh, (short)0, acc[mi][nj], false, false);
            }
        }
    }

    const int rsel = (lane & 16) ? 8 : 0;
#pragma unroll
    for (int nj = 0; nj < 4; ++nj) {
        int gc = c0 + (wn * 4 + nj) * 16 + (lane & 15);
        float bv = bias[gc];
#pragma unroll
        for (int mi = 0; mi < 2; ++mi) {
            int rb = n0 + (wm * 2 + mi) * 16 + rsel;
#pragma unroll
            for (int v = 0; v < 8; ++v) {
                int gr = rb + v;
                float yp = acc[mi][nj][v] + bv;
                float xm = X[(size_t)gr * D + gc];
                out[(size_t)gr * D + gc] = (xm != 0.0f) ? yp : 0.0f;
            }
        }
    }
}

// One block per row: sum masked row (masked entries already 0), normalize in place.
__global__ __launch_bounds__(256) void rownorm_kernel(float* __restrict__ out, int D)
{
    __shared__ float red[256];
    float* p = out + (size_t)blockIdx.x * D;
    float s = 0.f;
    for (int i = threadIdx.x; i < D; i += 256) s += p[i];
    red[threadIdx.x] = s;
    __syncthreads();
#pragma unroll
    for (int off = 128; off > 0; off >>= 1) {
        if (threadIdx.x < off) red[threadIdx.x] += red[threadIdx.x + off];
        __syncthreads();
    }
    float rs = red[0];
    for (int i = threadIdx.x; i < D; i += 256) {
        float v = p[i];
        p[i] = (rs != 0.f) ? (v / rs) : 0.f;
    }
}

extern "C" void kernel_launch(void* const* d_in, const int* in_sizes, int n_in,
                              void* d_out, int out_size, void* d_ws, size_t ws_size,
                              hipStream_t stream) {
    (void)n_in; (void)out_size;
    const float* X = (const float*)d_in[0];
    const float* A = (const float*)d_in[1];
    const float* b = (const float*)d_in[2];
    float* out = (float*)d_out;

    const int D = in_sizes[2];
    const int N = in_sizes[0] / D;
    const size_t nX = (size_t)N * D;
    const size_t nA = (size_t)D * D;
    const size_t need = (2 * nX + 2 * nA) * sizeof(unsigned short);

    dim3 grid(D / TILE, N / TILE);

    if (ws_size >= need) {
        unsigned short* Xh = (unsigned short*)d_ws;
        unsigned short* Xl = Xh + nX;
        unsigned short* Ah = Xl + nX;
        unsigned short* Al = Ah + nA;
        split_kernel<<<(unsigned)(nX / 1024), 256, 0, stream>>>(X, Xh, Xl, (long)(nX / 4));
        split_kernel<<<(unsigned)(nA / 1024), 256, 0, stream>>>(A, Ah, Al, (long)(nA / 4));
        gemm_mask_pre_kernel<<<grid, 256, 0, stream>>>(Xh, Xl, Ah, Al, X, b, out, N, D);
    } else {
        gemm_mask_conv_kernel<<<grid, 256, 0, stream>>>(X, A, b, out, N, D);
    }
    rownorm_kernel<<<N, 256, 0, stream>>>(out, D);
}
